// Model_47588237639844
// MI455X (gfx1250) — matrix operand order repaired and checked by reading, not compiled, not run on hardware
//
#include <hip/hip_runtime.h>

typedef __attribute__((ext_vector_type(16))) _Float16 v16h;
typedef __attribute__((ext_vector_type(8)))  _Float16 v8h;
typedef __attribute__((ext_vector_type(8)))  float    v8f;
typedef __attribute__((ext_vector_type(4)))  float    v4f;

#define CRF_B 1024
#define CRF_T 512
#define CRF_K 64

#define LOG2E 1.4426950408889634f
#define LN2   0.6931471805599453f

// ---------------------------------------------------------------------------
// Kernel 1: unary + pairwise gold-path scores. One thread per (b,t).
// Deterministic: block-tree reduce, one partial per block.
// ---------------------------------------------------------------------------
__global__ __launch_bounds__(256) void crf_emit(
    const float* __restrict__ logits, const float* __restrict__ trans,
    const int* __restrict__ gold, const int* __restrict__ seq_len,
    float* __restrict__ partial)
{
    int idx = blockIdx.x * 256 + threadIdx.x;   // idx = b*T + t, exactly B*T threads
    int b = idx >> 9;                            // / 512
    int t = idx & 511;
    int sl = seq_len[b];
    int g  = gold[idx];
    float v = 0.0f;
    if (t < sl) v += logits[(size_t)idx * CRF_K + g];
    if ((t + 1 < CRF_T) && (t < sl - 1)) v += trans[g * CRF_K + gold[idx + 1]];

    __shared__ float red[256];
    red[threadIdx.x] = v;
    __syncthreads();
#pragma unroll
    for (int s = 128; s > 0; s >>= 1) {
        if ((int)threadIdx.x < s) red[threadIdx.x] += red[threadIdx.x + s];
        __syncthreads();
    }
    if (threadIdx.x == 0) partial[blockIdx.x] = red[0];
}

// ---------------------------------------------------------------------------
// Kernel 2: CRF forward recursion via WMMA, alpha kept in log2 domain.
// One wave per block; each wave owns 16 batch rows.
// alpha C/D layout: M = state (64 = 4 tiles), N = batch (16):
//   v8f al[m], elem e, lane l:  state j = 16m + e + 8*(l>>4), batch = l&15.
// Step: D = A x B, A = exp(trans[j][i]) (f16 A-layout, registers),
// B = exp2(al2 - mx2) (f16 B-layout via 2KB LDS roundtrip), f32 accumulate;
// al2' = log2(D) + mx2 + logits*LOG2E, frozen by FMA blend where t >= seq_len.
// ---------------------------------------------------------------------------
__global__ __launch_bounds__(32) void crf_forward(
    const float* __restrict__ logits, const float* __restrict__ trans,
    const int* __restrict__ seq_len, float* __restrict__ third /* [B] */)
{
    __shared__ alignas(32) _Float16 sexp[16 * CRF_K];  // batch-major: [batch][state]

    const int lane = threadIdx.x;
    const int ln   = lane & 15;     // batch column within tile / A-row within tile
    const int h    = lane >> 4;     // lane-half
    const int b    = blockIdx.x * 16 + ln;

    // Build E = exp(trans[j][i]) in the 16-bit A-matrix 16x32 register layout.
    // A VGPR q (v16h elems 2q,2q+1), lane-half h: K = 32c + 8h + 2q + (q>=4 ? 8 : 0).
    v16h ea[2][4];
#pragma unroll
    for (int c = 0; c < 2; ++c)
#pragma unroll
        for (int m = 0; m < 4; ++m) {
            const float* tr = trans + (16 * m + ln) * CRF_K + 32 * c + 8 * h;
#pragma unroll
            for (int q = 0; q < 8; ++q) {
                int kb = 2 * q + (q >= 4 ? 8 : 0);
                ea[c][m][2 * q]     = (_Float16)__builtin_amdgcn_exp2f(tr[kb] * LOG2E);
                ea[c][m][2 * q + 1] = (_Float16)__builtin_amdgcn_exp2f(tr[kb + 1] * LOG2E);
            }
        }

    const int sl = seq_len[b];
    int ml = sl;                                   // wave-max seq_len (early exit)
#pragma unroll
    for (int o = 1; o < 32; o <<= 1) {
        int other = __shfl_xor(ml, o, 32);
        ml = ml > other ? ml : other;
    }

    const v4f* lgp = (const v4f*)(logits + (size_t)b * CRF_T * CRF_K);

    // alpha init = logits[:, 0, :] * LOG2E (log2 domain) in C-layout
    v8f al[4];
#pragma unroll
    for (int m = 0; m < 4; ++m) {
        v4f lo = lgp[4 * m + 2 * h];
        v4f hi = lgp[4 * m + 2 * h + 1];
#pragma unroll
        for (int e = 0; e < 4; ++e) {
            al[m][e]     = lo[e] * LOG2E;
            al[m][e + 4] = hi[e] * LOG2E;
        }
    }

    for (int t = 1; t < ml; ++t) {
        // Issue logits loads for this step early (off the critical chain).
        v4f lg0[4], lg1[4];
#pragma unroll
        for (int m = 0; m < 4; ++m) {
            lg0[m] = lgp[t * 16 + 4 * m + 2 * h];
            lg1[m] = lgp[t * 16 + 4 * m + 2 * h + 1];
        }

        // Per-batch running max (tree, depth ~5, then one half-swap).
        float pm[4];
#pragma unroll
        for (int m = 0; m < 4; ++m) {
            float a01 = fmaxf(al[m][0], al[m][1]);
            float a23 = fmaxf(al[m][2], al[m][3]);
            float a45 = fmaxf(al[m][4], al[m][5]);
            float a67 = fmaxf(al[m][6], al[m][7]);
            pm[m] = fmaxf(fmaxf(a01, a23), fmaxf(a45, a67));
        }
        float mx = fmaxf(fmaxf(pm[0], pm[1]), fmaxf(pm[2], pm[3]));
        mx = fmaxf(mx, __shfl_xor(mx, 16, 32));

        // exp2(al - mx) -> f16, store batch-major to LDS (ds_store_b128 x4).
#pragma unroll
        for (int m = 0; m < 4; ++m) {
            v8h p;
#pragma unroll
            for (int e = 0; e < 8; ++e)
                p[e] = (_Float16)__builtin_amdgcn_exp2f(al[m][e] - mx);
            *(v8h*)&sexp[ln * CRF_K + 16 * m + 8 * h] = p;
        }

        // Reload in B-matrix 32x16 layout (ds_load_b128 x2 per chunk) and WMMA.
        v8f acc[4] = {};
#pragma unroll
        for (int c = 0; c < 2; ++c) {
            v8h b0 = *(const v8h*)&sexp[ln * CRF_K + 32 * c + 8 * h];
            v8h b1 = *(const v8h*)&sexp[ln * CRF_K + 32 * c + 16 + 8 * h];
            v16h bv = __builtin_shufflevector(b0, b1, 0, 1, 2, 3, 4, 5, 6, 7, 8, 9, 10, 11, 12, 13, 14, 15);
#pragma unroll
            for (int m = 0; m < 4; ++m)
                acc[m] = __builtin_amdgcn_wmma_f32_16x16x32_f16(
                    false, ea[c][m], false, bv, (short)0, acc[m], false, false);
        }

        // al' = log2(D) + mx + lg*LOG2E; freeze via branch-proof FMA blend.
        const float fmask = (t < sl) ? 1.0f : 0.0f;
#pragma unroll
        for (int m = 0; m < 4; ++m)
#pragma unroll
            for (int e = 0; e < 8; ++e) {
                float lgv = (e < 4) ? lg0[m][e] : lg1[m][e - 4];
                float base = __builtin_fmaf(lgv, LOG2E, mx);          // mx + lg*log2e
                float na   = __builtin_amdgcn_logf(acc[m][e]) + base; // v_log_f32
                al[m][e]   = __builtin_fmaf(fmask, na - al[m][e], al[m][e]);
            }
    }

    // third[b] = LN2 * (mx2 + log2(sum exp2(al - mx2)))  (natural-log result)
    float pm[4];
#pragma unroll
    for (int m = 0; m < 4; ++m) {
        float a01 = fmaxf(al[m][0], al[m][1]);
        float a23 = fmaxf(al[m][2], al[m][3]);
        float a45 = fmaxf(al[m][4], al[m][5]);
        float a67 = fmaxf(al[m][6], al[m][7]);
        pm[m] = fmaxf(fmaxf(a01, a23), fmaxf(a45, a67));
    }
    float mx = fmaxf(fmaxf(pm[0], pm[1]), fmaxf(pm[2], pm[3]));
    mx = fmaxf(mx, __shfl_xor(mx, 16, 32));

    float ps[4];
#pragma unroll
    for (int m = 0; m < 4; ++m) {
        float e0 = __builtin_amdgcn_exp2f(al[m][0] - mx);
        float e1 = __builtin_amdgcn_exp2f(al[m][1] - mx);
        float e2 = __builtin_amdgcn_exp2f(al[m][2] - mx);
        float e3 = __builtin_amdgcn_exp2f(al[m][3] - mx);
        float e4 = __builtin_amdgcn_exp2f(al[m][4] - mx);
        float e5 = __builtin_amdgcn_exp2f(al[m][5] - mx);
        float e6 = __builtin_amdgcn_exp2f(al[m][6] - mx);
        float e7 = __builtin_amdgcn_exp2f(al[m][7] - mx);
        ps[m] = ((e0 + e1) + (e2 + e3)) + ((e4 + e5) + (e6 + e7));
    }
    float s = (ps[0] + ps[1]) + (ps[2] + ps[3]);
    s += __shfl_xor(s, 16, 32);
    if (h == 0) third[b] = LN2 * (mx + __builtin_amdgcn_logf(s));
}

// ---------------------------------------------------------------------------
// Kernel 3: deterministic final reduce.  out = (third - (first+second)) / B
// ---------------------------------------------------------------------------
__global__ __launch_bounds__(256) void crf_finalize(
    const float* __restrict__ pe,   // 2048 emit partials
    const float* __restrict__ pt,   // 1024 per-batch third values
    float* __restrict__ out)
{
    __shared__ float red[256];
    float v = 0.0f;
    for (int i = threadIdx.x; i < 2048; i += 256) v -= pe[i];
    for (int i = threadIdx.x; i < 1024; i += 256) v += pt[i];
    red[threadIdx.x] = v;
    __syncthreads();
#pragma unroll
    for (int s = 128; s > 0; s >>= 1) {
        if ((int)threadIdx.x < s) red[threadIdx.x] += red[threadIdx.x + s];
        __syncthreads();
    }
    if (threadIdx.x == 0) out[0] = red[0] * (1.0f / CRF_B);
}

// ---------------------------------------------------------------------------
extern "C" void kernel_launch(void* const* d_in, const int* in_sizes, int n_in,
                              void* d_out, int out_size, void* d_ws, size_t ws_size,
                              hipStream_t stream)
{
    const float* logits = (const float*)d_in[0];   // (B,T,K) f32
    const float* trans  = (const float*)d_in[1];   // (K,K)   f32
    const int*   gold   = (const int*)d_in[2];     // (B,T)   i32
    const int*   seqlen = (const int*)d_in[3];     // (B,)    i32
    float* out = (float*)d_out;

    float* pe = (float*)d_ws;        // 2048 floats: emit partials
    float* pt = pe + 2048;           // 1024 floats: per-batch logsumexp

    crf_emit<<<(CRF_B * CRF_T) / 256, 256, 0, stream>>>(logits, trans, gold, seqlen, pe);
    crf_forward<<<CRF_B / 16, 32, 0, stream>>>(logits, trans, seqlen, pt);
    crf_finalize<<<1, 256, 0, stream>>>(pe, pt, out);
}
